// MultiheadSelfAttention_61555471286489
// MI455X (gfx1250) — matrix operand order repaired and checked by reading, not compiled, not run on hardware
//
#include <hip/hip_runtime.h>
#include <hip/hip_bf16.h>

// ---------------------------------------------------------------------------
// Multi-head self-attention forward for MI455X (gfx1250, wave32, WMMA).
//   cvt(x,Wqkv,Wout)->bf16 ; QKV GEMM ; FUSED(scores+softmax+attn@V) ;
//   out = values@Wout + b.
// All GEMMs: v_wmma_f32_16x16x32_bf16, f32 accumulation.
// d_out = [ out: 2*2048*1024 f32 | attention: 2*16*2048*2048 f32 ]
// Attention tensor written exactly once (normalized); softmax stats online in
// registers; scores recomputed in pass 2 (FLOPs cheap, HBM not).
// Global->LDS staging uses CDNA5 async-to-LDS (ASYNCcnt); pass 1 of the fused
// kernel ping-pongs K tiles through sKT/sV with register-pipelined staging.
// ---------------------------------------------------------------------------

typedef __attribute__((ext_vector_type(16))) __bf16 v16bf;
typedef __attribute__((ext_vector_type(8)))  float  v8f;
typedef __attribute__((ext_vector_type(4)))  int    v4i;

#define DIM       1024
#define SEQ       2048
#define BATCH     2
#define HEADS     16
#define HDIM      64
#define TOKENS    (BATCH * SEQ)      // 4096
#define QKV_W     (3 * DIM)          // 3072

#define AP 40    // gemm sA pitch (elems)
#define BP 72    // gemm sB pitch / fused 72-pitch tiles

union FragBF16 { uint4 u4[2]; v16bf v; };

__device__ __forceinline__ v8f wmma_bf16(const FragBF16& a, const FragBF16& b, v8f c) {
    return __builtin_amdgcn_wmma_f32_16x16x32_bf16(false, a.v, false, b.v,
                                                   (short)0, c, false, false);
}

// ---------------- async global->LDS copy (CDNA5), with safe fallback --------
#if defined(__has_builtin)
#  if __has_builtin(__builtin_amdgcn_global_load_async_to_lds_b128) && \
      __has_builtin(__builtin_amdgcn_s_wait_asynccnt)
#    define HAVE_ASYNC_LDS 1
#  endif
#endif
#ifndef HAVE_ASYNC_LDS
#  define HAVE_ASYNC_LDS 0
#endif

#if HAVE_ASYNC_LDS
typedef __attribute__((address_space(1))) v4i* gv4i_p;   // global int4*
typedef __attribute__((address_space(3))) v4i* lv4i_p;   // LDS int4*
#endif

__device__ __forceinline__ void cp_async_b128(void* dstLds, const void* srcGlobal) {
#if HAVE_ASYNC_LDS
    __builtin_amdgcn_global_load_async_to_lds_b128(
        (gv4i_p)srcGlobal, (lv4i_p)dstLds, 0, 0);
#else
    *(uint4*)dstLds = *(const uint4*)srcGlobal;
#endif
}
__device__ __forceinline__ void cp_async_wait() {
#if HAVE_ASYNC_LDS
    __builtin_amdgcn_s_wait_asynccnt(0);
#endif
}

// ---------------------------------------------------------------------------
// f32 -> bf16 converter
// ---------------------------------------------------------------------------
__global__ void __launch_bounds__(256)
mhsa_cvt_f32_bf16(const float* __restrict__ in, __bf16* __restrict__ out, int n) {
    int i = (blockIdx.x * 256 + threadIdx.x) * 4;
    if (i + 3 < n) {
        float4 f = *(const float4*)(in + i);
        __bf16 h[4];
        h[0] = (__bf16)f.x; h[1] = (__bf16)f.y;
        h[2] = (__bf16)f.z; h[3] = (__bf16)f.w;
        *(uint2*)(out + i) = *(const uint2*)h;
    }
}

// ---------------------------------------------------------------------------
// Projection GEMM: C = A[bf16] @ B[bf16] (+bias).  Tile 128x64, 8 waves,
// wave = 32x32 patch (2x2 wmma).  Double-buffered LDS with async staging.
// ---------------------------------------------------------------------------
template<bool OUT_BF16, bool HAS_BIAS>
__device__ __forceinline__ void gemm_block(
    const __bf16* __restrict__ A, int lda,
    const __bf16* __restrict__ B, int ldb,
    const float* __restrict__ bias,
    void* __restrict__ Cptr, int ldc,
    int Kdim, int m0, int n0)
{
    __shared__ __align__(16) __bf16 sA[2 * 128 * AP];
    __shared__ __align__(16) __bf16 sB[2 * 32 * BP];

    const int tid  = threadIdx.x;
    const int lane = tid & 31;
    const int wave = tid >> 5;
    const int wm = (wave >> 1) * 32;
    const int wn = (wave & 1)  * 32;

    v8f acc[2][2] = {};

    const int a_m  = lane & 15;
    const int a_kg = (lane >> 4) * 8;
    const int b_k  = (lane >> 4) * 8;

    auto stage = [&](int k0, int buf) {
        for (int c = tid; c < 512; c += 256) {
            int row = c >> 2, col = (c & 3) * 8;
            cp_async_b128(&sA[buf * 128 * AP + row * AP + col],
                          A + (size_t)(m0 + row) * lda + k0 + col);
        }
        {
            int c = tid; int row = c >> 3, col = (c & 7) * 8;
            cp_async_b128(&sB[buf * 32 * BP + row * BP + col],
                          B + (size_t)(k0 + row) * ldb + n0 + col);
        }
    };

    stage(0, 0);
    cp_async_wait();
    __syncthreads();

    int buf = 0;
    for (int k0 = 0; k0 < Kdim; k0 += 32) {
        if (k0 + 32 < Kdim) stage(k0 + 32, buf ^ 1);   // prefetch next tile

        const __bf16* bA = &sA[buf * 128 * AP];
        const __bf16* bB = &sB[buf * 32 * BP];

        FragBF16 af[2], bfr[2];
        #pragma unroll
        for (int mi = 0; mi < 2; ++mi) {
            const __bf16* p = &bA[(wm + mi * 16 + a_m) * AP + a_kg];
            af[mi].u4[0] = *(const uint4*)p;
            af[mi].u4[1] = *(const uint4*)(p + 16);
        }
        #pragma unroll
        for (int ni = 0; ni < 2; ++ni) {
            const __bf16* p = &bB[b_k * BP + wn + ni * 16 + (lane & 15)];
            for (int s = 0; s < 8; ++s) { bfr[ni].v[s] = p[s * BP]; bfr[ni].v[8 + s] = p[(16 + s) * BP]; }
        }
        #pragma unroll
        for (int mi = 0; mi < 2; ++mi)
            #pragma unroll
            for (int ni = 0; ni < 2; ++ni)
                acc[mi][ni] = wmma_bf16(af[mi], bfr[ni], acc[mi][ni]);

        cp_async_wait();
        __syncthreads();
        buf ^= 1;
    }

    const int cn   = lane & 15;
    const int rsel = (lane >> 4) * 8;
    #pragma unroll
    for (int mi = 0; mi < 2; ++mi) {
        #pragma unroll
        for (int ni = 0; ni < 2; ++ni) {
            int col = n0 + wn + ni * 16 + cn;
            float bv = HAS_BIAS ? bias[col] : 0.0f;
            #pragma unroll
            for (int r = 0; r < 8; ++r) {
                int row = m0 + wm + mi * 16 + rsel + r;
                float v = acc[mi][ni][r] + bv;
                if (OUT_BF16)
                    ((__bf16*)Cptr)[(size_t)row * ldc + col] = (__bf16)v;
                else
                    ((float*)Cptr)[(size_t)row * ldc + col] = v;
            }
        }
    }
}

__global__ void __launch_bounds__(256)
mhsa_gemm_qkv(const __bf16* __restrict__ A, const __bf16* __restrict__ B,
              const float* __restrict__ bias, __bf16* __restrict__ C) {
    gemm_block<true, true>(A, DIM, B, QKV_W, bias, C, QKV_W,
                           DIM, blockIdx.y * 128, blockIdx.x * 64);
}

__global__ void __launch_bounds__(256)
mhsa_gemm_out(const __bf16* __restrict__ A, const __bf16* __restrict__ B,
              const float* __restrict__ bias, float* __restrict__ C) {
    gemm_block<false, true>(A, DIM, B, DIM, bias, C, DIM,
                            DIM, blockIdx.y * 128, blockIdx.x * 64);
}

// ---------------------------------------------------------------------------
// Fused attention: per (b,h) and 128-row q-block.
//   8 waves; wave owns 16 q rows across the full k extent (stats wave-private).
//   Pass 1: online softmax stats; K tiles ping-pong through sKT/sV with
//           register-pipelined staging (global load overlaps WMMA+stats).
//   Pass 2: recompute scores, write normalized attention once, accumulate p@V.
// ---------------------------------------------------------------------------
__global__ void __launch_bounds__(256)
mhsa_fused_attn(const __bf16* __restrict__ qkv, float* __restrict__ attn,
                __bf16* __restrict__ vals)
{
    __shared__ __align__(16) __bf16 sQ [128 * BP];      // [qrow][d]
    __shared__ __align__(16) __bf16 sKT[64  * BP];      // [d][kidx] (K^T) / pass1 buf0
    __shared__ __align__(16) __bf16 sV [64  * BP];      // [kidx][d] (V)   / pass1 buf1
    __shared__ __align__(16) __bf16 sP [8 * 16 * BP];   // per-wave p tile [16][BP]

    const int tid  = threadIdx.x;
    const int lane = tid & 31;
    const int wave = tid >> 5;
    const int q0   = blockIdx.x * 128;
    const int bh   = blockIdx.y;
    const int b    = bh >> 4, h = bh & 15;

    const __bf16* Qg = qkv + (size_t)b * SEQ * QKV_W + h * (3 * HDIM);
    const __bf16* Kg = Qg + HDIM;
    const __bf16* Vg = Qg + 2 * HDIM;
    float* attnBH = attn + (size_t)bh * SEQ * SEQ;

    const int wm   = wave * 16;              // wave's q-row base within block
    const int a_m  = lane & 15;
    const int a_kg = (lane >> 4) * 8;
    const int b_k  = (lane >> 4) * 8;
    const int cn   = lane & 15;
    const int rsel = (lane >> 4) * 8;
    __bf16* sPw = &sP[wave * 16 * BP];

    // this thread's two staging chunks of a 128(kidx) x 32(d) K block (uint4 = 8 bf16 of d)
    const int kr0 = tid >> 2,          kd0 = (tid & 3) * 8;          // chunk 0
    const int kr1 = (tid + 256) >> 2,  kd1 = ((tid + 256) & 3) * 8;  // chunk 1

    // ---- stage Q block once: 128 x 64 (async)
    for (int c = tid; c < 1024; c += 256) {
        int row = c >> 3, col = (c & 7) * 8;
        cp_async_b128(&sQ[row * BP + col], Qg + (size_t)(q0 + row) * QKV_W + col);
    }
    cp_async_wait();
    __syncthreads();

    // wave-constant Q fragments (16 rows x 64 d = 2 K-steps)
    FragBF16 aQ[2];
    #pragma unroll
    for (int ks = 0; ks < 2; ++ks) {
        const __bf16* p = &sQ[(wm + a_m) * BP + ks * 32 + a_kg];
        aQ[ks].u4[0] = *(const uint4*)p;
        aQ[ks].u4[1] = *(const uint4*)(p + 16);
    }

    float mrun[8], srun[8];
    #pragma unroll
    for (int r = 0; r < 8; ++r) { mrun[r] = -3.402823466e38f; srun[r] = 0.0f; }

    // =============================== PASS 1 ================================
    // K tiles ping-pong between sKT (buf 0) and sV (buf 1); both hold K^T
    // [d][kidx].  Stage tile 0 into buf 0, then pipeline.
    {
        uint4 u0 = *(const uint4*)(Kg + (size_t)kr0 * QKV_W + kd0);
        uint4 u1 = *(const uint4*)(Kg + (size_t)kr1 * QKV_W + kd1);
        const __bf16* e0 = (const __bf16*)&u0;
        const __bf16* e1 = (const __bf16*)&u1;
        #pragma unroll
        for (int j = 0; j < 8; ++j) sKT[(kd0 + j) * BP + kr0] = e0[j];
        #pragma unroll
        for (int j = 0; j < 8; ++j) sKT[(kd1 + j) * BP + kr1] = e1[j];
    }
    __syncthreads();

    int cur = 0;
    for (int it = 0; it < SEQ / 64; ++it) {
        // issue next tile's global loads into registers (wait lands at scatter)
        uint4 u0, u1;
        const bool nxt = (it + 1) < (SEQ / 64);
        if (nxt) {
            const __bf16* Kn = Kg + (size_t)(it * 64 + 64) * QKV_W;
            u0 = *(const uint4*)(Kn + (size_t)kr0 * QKV_W + kd0);
            u1 = *(const uint4*)(Kn + (size_t)kr1 * QKV_W + kd1);
        }

        const __bf16* K = cur ? sV : sKT;

        // scores: wave tile 16(q) x 64(kidx) = 4 subtiles x 2 K-steps
        v8f sc[4] = {};
        #pragma unroll
        for (int ks = 0; ks < 2; ++ks)
            #pragma unroll
            for (int ns = 0; ns < 4; ++ns) {
                FragBF16 bK;
                const __bf16* p = &K[(ks * 32 + b_k) * BP + ns * 16 + cn];
                for (int s = 0; s < 8; ++s) { bK.v[s] = p[s * BP]; bK.v[8 + s] = p[(16 + s) * BP]; }
                sc[ns] = wmma_bf16(aQ[ks], bK, sc[ns]);
            }
        #pragma unroll
        for (int ns = 0; ns < 4; ++ns) sc[ns] = sc[ns] * 0.125f;

        // online softmax stats (per row; rows are wave-private)
        #pragma unroll
        for (int r = 0; r < 8; ++r) {
            float v = fmaxf(fmaxf(sc[0][r], sc[1][r]), fmaxf(sc[2][r], sc[3][r]));
            v = fmaxf(v, __shfl_xor(v, 1, 32));
            v = fmaxf(v, __shfl_xor(v, 2, 32));
            v = fmaxf(v, __shfl_xor(v, 4, 32));
            v = fmaxf(v, __shfl_xor(v, 8, 32));
            float mnew = fmaxf(mrun[r], v);
            float t = __expf(sc[0][r] - mnew) + __expf(sc[1][r] - mnew)
                    + __expf(sc[2][r] - mnew) + __expf(sc[3][r] - mnew);
            t += __shfl_xor(t, 1, 32);
            t += __shfl_xor(t, 2, 32);
            t += __shfl_xor(t, 4, 32);
            t += __shfl_xor(t, 8, 32);
            srun[r] = srun[r] * __expf(mrun[r] - mnew) + t;
            mrun[r] = mnew;
        }

        // scatter next tile into the other buffer (transposed)
        if (nxt) {
            __bf16* D = cur ? sKT : sV;
            const __bf16* e0 = (const __bf16*)&u0;
            const __bf16* e1 = (const __bf16*)&u1;
            #pragma unroll
            for (int j = 0; j < 8; ++j) D[(kd0 + j) * BP + kr0] = e0[j];
            #pragma unroll
            for (int j = 0; j < 8; ++j) D[(kd1 + j) * BP + kr1] = e1[j];
        }
        __syncthreads();
        cur ^= 1;
    }

    float inv[8];
    #pragma unroll
    for (int r = 0; r < 8; ++r) inv[r] = 1.0f / srun[r];

    v8f accv[4] = {};

    // =============================== PASS 2 ================================
    for (int it = 0; it < SEQ / 64; ++it) {
        const int kb = it * 64;
        __syncthreads();                       // previous tiles fully consumed
        // V tile streams via the async engine while we transpose K
        for (int c = tid; c < 512; c += 256) {
            int kr = c >> 2, d0 = (c & 3) * 8;
            cp_async_b128(&sV[kr * BP + d0], Vg + (size_t)(kb + kr) * QKV_W + d0);
        }
        {
            const __bf16* Kt = Kg + (size_t)kb * QKV_W;
            uint4 u0 = *(const uint4*)(Kt + (size_t)kr0 * QKV_W + kd0);
            uint4 u1 = *(const uint4*)(Kt + (size_t)kr1 * QKV_W + kd1);
            const __bf16* e0 = (const __bf16*)&u0;
            const __bf16* e1 = (const __bf16*)&u1;
            #pragma unroll
            for (int j = 0; j < 8; ++j) sKT[(kd0 + j) * BP + kr0] = e0[j];
            #pragma unroll
            for (int j = 0; j < 8; ++j) sKT[(kd1 + j) * BP + kr1] = e1[j];
        }
        cp_async_wait();
        __syncthreads();

        // recompute scores
        v8f sc[4] = {};
        #pragma unroll
        for (int ks = 0; ks < 2; ++ks)
            #pragma unroll
            for (int ns = 0; ns < 4; ++ns) {
                FragBF16 bK;
                const __bf16* p = &sKT[(ks * 32 + b_k) * BP + ns * 16 + cn];
                for (int s = 0; s < 8; ++s) { bK.v[s] = p[s * BP]; bK.v[8 + s] = p[(16 + s) * BP]; }
                sc[ns] = wmma_bf16(aQ[ks], bK, sc[ns]);
            }

        // p = softmax value; write attention (only write of this tensor);
        // stage p into wave-private LDS for the A-side of p@V
        #pragma unroll
        for (int ns = 0; ns < 4; ++ns) {
            #pragma unroll
            for (int r = 0; r < 8; ++r) {
                float p = __expf(sc[ns][r] * 0.125f - mrun[r]) * inv[r];
                int row = q0 + wm + rsel + r;
                attnBH[(size_t)row * SEQ + kb + ns * 16 + cn] = p;
                sPw[(rsel + r) * BP + ns * 16 + cn] = (__bf16)p;
            }
        }

        // p @ V : wave tile 16(q) x 64(hd), K = 64 (2 K-steps)
        FragBF16 pf[2];
        #pragma unroll
        for (int ks = 0; ks < 2; ++ks) {
            const __bf16* p = &sPw[a_m * BP + ks * 32 + a_kg];
            pf[ks].u4[0] = *(const uint4*)p;
            pf[ks].u4[1] = *(const uint4*)(p + 16);
        }
        #pragma unroll
        for (int ks = 0; ks < 2; ++ks)
            #pragma unroll
            for (int nv = 0; nv < 4; ++nv) {
                FragBF16 bV;
                const __bf16* p = &sV[(ks * 32 + b_k) * BP + nv * 16 + cn];
                for (int s = 0; s < 8; ++s) { bV.v[s] = p[s * BP]; bV.v[8 + s] = p[(16 + s) * BP]; }
                accv[nv] = wmma_bf16(pf[ks], bV, accv[nv]);
            }
    }

    // ---- write values tile: rows q0+wm.. , cols h*64..  (bf16)
    __bf16* Vout = vals + (size_t)b * SEQ * DIM + h * HDIM;
    #pragma unroll
    for (int nv = 0; nv < 4; ++nv)
        #pragma unroll
        for (int r = 0; r < 8; ++r) {
            int row = q0 + wm + rsel + r;
            Vout[(size_t)row * DIM + nv * 16 + cn] = (__bf16)accv[nv][r];
        }
}

// ---------------------------------------------------------------------------
// Host launcher
// ---------------------------------------------------------------------------
extern "C" void kernel_launch(void* const* d_in, const int* in_sizes, int n_in,
                              void* d_out, int out_size, void* d_ws, size_t ws_size,
                              hipStream_t stream) {
    const float* x     = (const float*)d_in[0];
    const float* W_qkv = (const float*)d_in[1];
    const float* b_qkv = (const float*)d_in[2];
    const float* W_out = (const float*)d_in[3];
    const float* b_out = (const float*)d_in[4];

    float* out  = (float*)d_out;
    float* attn = out + (size_t)TOKENS * DIM;

    char* ws = (char*)d_ws;
    __bf16* xb    = (__bf16*)ws;  ws += (size_t)TOKENS * DIM   * 2;
    __bf16* wqkvb = (__bf16*)ws;  ws += (size_t)DIM    * QKV_W * 2;
    __bf16* wob   = (__bf16*)ws;  ws += (size_t)DIM    * DIM   * 2;
    __bf16* qkvb  = (__bf16*)ws;  ws += (size_t)TOKENS * QKV_W * 2;
    __bf16* valb  = (__bf16*)ws;

    mhsa_cvt_f32_bf16<<<(TOKENS * DIM / 4) / 256, 256, 0, stream>>>(x, xb, TOKENS * DIM);
    mhsa_cvt_f32_bf16<<<(DIM * QKV_W / 4) / 256, 256, 0, stream>>>(W_qkv, wqkvb, DIM * QKV_W);
    mhsa_cvt_f32_bf16<<<(DIM * DIM / 4) / 256, 256, 0, stream>>>(W_out, wob, DIM * DIM);

    mhsa_gemm_qkv<<<dim3(QKV_W / 64, TOKENS / 128), 256, 0, stream>>>(xb, wqkvb, b_qkv, qkvb);

    mhsa_fused_attn<<<dim3(SEQ / 128, BATCH * HEADS), 256, 0, stream>>>(qkvb, attn, valb);

    mhsa_gemm_out<<<dim3(DIM / 64, TOKENS / 128), 256, 0, stream>>>(valb, wob, b_out, out);
}
